// rosa_4bit_layer_7928509628835
// MI455X (gfx1250) — 8-bit matrix operand order repaired and checked by reading, compiled, not run on hardware
//
#include <hip/hip_runtime.h>
#include <stdint.h>

#define B_  8
#define T_  1024
#define C_  1024
#define CG_ 256

typedef __attribute__((ext_vector_type(8))) int v8i;

__global__ __launch_bounds__(256) void k_tokenize(const float* __restrict__ x,
                                                  uint8_t* __restrict__ tok) {
  __shared__ uint8_t tile[32][CG_ + 4];
  const int tid = threadIdx.x;
  const int b   = blockIdx.x >> 5;
  const int t0  = (blockIdx.x & 31) << 5;
  #pragma unroll
  for (int k = 0; k < 32; ++k) {
    const int l  = tid + (k << 8);
    const int tl = l >> 8;
    const int cg = l & 255;
    const float4 v = ((const float4*)x)[((size_t)(b * T_ + t0 + tl) * C_ + cg * 4) >> 2];
    const uint32_t tk = (v.x > 0.f ? 1u : 0u) | (v.y > 0.f ? 2u : 0u) |
                        (v.z > 0.f ? 4u : 0u) | (v.w > 0.f ? 8u : 0u);
    tile[tl][cg] = (uint8_t)tk;
  }
  __syncthreads();
  #pragma unroll
  for (int k = 0; k < 32; ++k) {
    const int m  = tid + (k << 8);
    const int cg = m >> 5;
    const int tl = m & 31;
    tok[((size_t)(b * CG_ + cg) << 10) + t0 + tl] = tile[tl][cg];
  }
}

__global__ __launch_bounds__(256) void k_rosa_dp(const uint8_t* __restrict__ tok,
                                                 uint8_t* __restrict__ pred) {
  __shared__ __attribute__((aligned(16))) uint8_t s_tok[8][1024];
  __shared__ uint32_t s_occ[8][16][32];
  __shared__ uint32_t s_best[8][1024];
  const int w    = threadIdx.x >> 5;
  const int lane = threadIdx.x & 31;
  const int row  = (blockIdx.x << 3) + w;
  const uint8_t* trow = tok + ((size_t)row << 10);
  __builtin_prefetch(trow + lane * 32, 0, 0);

#if defined(__gfx1250__) && __has_builtin(__builtin_amdgcn_global_load_async_to_lds_b128)
  {
    typedef int v4g __attribute__((vector_size(16)));
    typedef __attribute__((address_space(1))) v4g* gptr_t;
    typedef __attribute__((address_space(3))) v4g* lptr_t;
    gptr_t g = (gptr_t)(trow + (size_t)lane * 16);
    lptr_t l = (lptr_t)(&s_tok[w][lane * 16]);
    __builtin_amdgcn_global_load_async_to_lds_b128(g, l, 0, 0);
    __builtin_amdgcn_global_load_async_to_lds_b128(g + 32, l + 32, 0, 0);
  }
  #define ROSA_ASYNC_STAGE 1
#endif

  uint4 p0 = ((const uint4*)trow)[lane * 2 + 0];
  uint4 p1 = ((const uint4*)trow)[lane * 2 + 1];
#ifndef ROSA_ASYNC_STAGE
  ((uint4*)s_tok[w])[lane * 2 + 0] = p0;
  ((uint4*)s_tok[w])[lane * 2 + 1] = p1;
#endif

  uint32_t pk[8] = {p0.x, p0.y, p0.z, p0.w, p1.x, p1.y, p1.z, p1.w};
  uint32_t bp[4] = {0u, 0u, 0u, 0u};
  #pragma unroll
  for (int d = 0; d < 8; ++d) {
    #pragma unroll
    for (int b = 0; b < 4; ++b) {
      const uint32_t y   = (pk[d] >> b) & 0x01010101u;
      const uint32_t nib = (y | (y >> 7) | (y >> 14) | (y >> 21)) & 0xFu;
      bp[b] |= nib << (d * 4);
    }
  }
  const uint32_t nb0 = ~bp[0], nb1 = ~bp[1], nb2 = ~bp[2], nb3 = ~bp[3];
  #pragma unroll
  for (int v = 0; v < 16; ++v) {
    const uint32_t m = ((v & 1) ? bp[0] : nb0) & ((v & 2) ? bp[1] : nb1) &
                       ((v & 4) ? bp[2] : nb2) & ((v & 8) ? bp[3] : nb3);
    s_occ[w][v][lane] = m;
  }

  int L[32];
  #pragma unroll
  for (int s = 0; s < 32; ++s) L[s] = 0;
  const int laneBase = lane << 5;
  if (lane == 0) s_best[w][0] = 0;

#ifdef ROSA_ASYNC_STAGE
  asm volatile("s_wait_asynccnt 0x0" ::: "memory");
#endif

  for (int i = 1; i < T_; ++i) {
    const int tok_i = s_tok[w][i];
    int carry = __shfl_up(L[31], 1, 32);
    carry = lane ? carry : 0;
    const uint32_t eb = s_occ[w][tok_i][lane];
    const int ip = i >> 5, il = i & 31;
    const uint32_t valid = (lane < ip) ? 0xFFFFFFFFu
                          : ((lane == ip) ? ((1u << il) - 1u) : 0u);
    const uint32_t mb = eb & valid;
    int best = 0;
    #pragma unroll
    for (int s = 31; s >= 0; --s) {
      const int sm   = ((int)(mb << (31 - s))) >> 31;
      const int prev = s ? L[s - 1] : carry;
      L[s] = (prev + 1) & sm;
      const int key = (L[s] << 10) | (laneBase + s);
      best = max(best, key);
    }
    #pragma unroll
    for (int off = 16; off > 0; off >>= 1)
      best = max(best, __shfl_xor(best, off, 32));
    if (lane == 0) s_best[w][i] = (uint32_t)best;
  }

  uint32_t ow[8];
  #pragma unroll
  for (int d = 0; d < 8; ++d) {
    uint32_t acc = 0;
    #pragma unroll
    for (int b4 = 0; b4 < 4; ++b4) {
      const int idx = laneBase + d * 4 + b4;
      const uint32_t v = s_best[w][idx];
      const uint32_t p = (v >> 10) ? (uint32_t)s_tok[w][(v & 1023u) + 1] : 0u;
      acc |= p << (b4 * 8);
    }
    ow[d] = acc;
  }
  uint8_t* prow = pred + ((size_t)row << 10);
  uint4 q0; q0.x = ow[0]; q0.y = ow[1]; q0.z = ow[2]; q0.w = ow[3];
  uint4 q1; q1.x = ow[4]; q1.y = ow[5]; q1.z = ow[6]; q1.w = ow[7];
  ((uint4*)prow)[lane * 2 + 0] = q0;
  ((uint4*)prow)[lane * 2 + 1] = q1;
}

__global__ __launch_bounds__(256) void k_expand(const uint8_t* __restrict__ pred,
                                                const float* __restrict__ emb0,
                                                const float* __restrict__ emb1,
                                                float* __restrict__ out) {
  const int lane = threadIdx.x & 31;
  const int wid  = (blockIdx.x << 3) + (threadIdx.x >> 5);
  const int cq = wid & 63;
  const int tb = (wid >> 6) & 63;
  const int b  = wid >> 12;
  const int t0 = tb << 4;
  const int cbase = cq << 4;
  const int cg0   = cq << 2;
  const int n    = lane & 15;
  const int half = lane >> 4;
  const int m15  = lane & 15;

  int tq[4];
  #pragma unroll
  for (int q = 0; q < 4; ++q)
    tq[q] = pred[(((size_t)(b * CG_ + cg0 + q)) << 10) + t0 + m15];
  v8i A;
  #pragma unroll
  for (int d = 0; d < 8; ++d) {
    const int off = tq[d >> 1] - ((d & 1) * 4 + half * 8);
    A[d] = (off >= 0 && off < 4) ? (int)(1u << ((off & 3) * 8)) : 0;
  }

  v8i Bm;
  #pragma unroll
  for (int d = 0; d < 8; ++d) {
    uint32_t acc = 0;
    #pragma unroll
    for (int byp = 0; byp < 4; ++byp) {
      const int K = (d >> 1) * 16 + half * 8 + (d & 1) * 4 + byp;
      const int q = K >> 4, v = K & 15;
      const uint32_t bit = (uint32_t)((q == (n >> 2)) && ((v >> (n & 3)) & 1));
      acc |= bit << (byp * 8);
    }
    Bm[d] = (int)acc;
  }

  v8i Cz = {0, 0, 0, 0, 0, 0, 0, 0};
  v8i D = __builtin_amdgcn_wmma_i32_16x16x64_iu8(false, A, false, Bm, Cz, false, false);

  const int c = cbase + n;
  const float e0 = emb0[c], e1 = emb1[c];
  #pragma unroll
  for (int r = 0; r < 8; ++r) {
    const int t = t0 + half * 8 + r;
    out[(size_t)(b * T_ + t) * C_ + c] = D[r] ? e1 : e0;
  }
}

extern "C" void kernel_launch(void* const* d_in, const int* in_sizes, int n_in,
                              void* d_out, int out_size, void* d_ws, size_t ws_size,
                              hipStream_t stream) {
  (void)in_sizes; (void)n_in; (void)out_size; (void)ws_size;
  const float* x  = (const float*)d_in[0];
  const float* e0 = (const float*)d_in[1];
  const float* e1 = (const float*)d_in[2];
  float* out = (float*)d_out;
  uint8_t* tok  = (uint8_t*)d_ws;
  uint8_t* pred = tok + (size_t)B_ * CG_ * T_;
  k_tokenize<<<dim3(256),  dim3(256), 0, stream>>>(x, tok);
  k_rosa_dp <<<dim3(256),  dim3(256), 0, stream>>>(tok, pred);
  k_expand  <<<dim3(4096), dim3(256), 0, stream>>>(pred, e0, e1, out);
}
